// VisionTransformer_81973745811985
// MI455X (gfx1250) — matrix operand order repaired and checked by reading, not compiled, not run on hardware
//
#include <hip/hip_runtime.h>
#include <hip/hip_bf16.h>

// ---------------- problem constants (ViT-Base, B=32) ----------------
#define B_     32
#define D_     768
#define L_     12
#define H_     12
#define DH_    64
#define F_     3072
#define NTOK   197
#define NPAD   208   // 13 * 16, padded sequence length for WMMA tiling
#define NPATCH 196

typedef unsigned short u16_t;
typedef __attribute__((ext_vector_type(16))) __bf16 v16bf;
typedef __attribute__((ext_vector_type(8)))  float  v8f;
typedef __attribute__((ext_vector_type(4)))  float  v4f;
typedef __attribute__((ext_vector_type(4)))  unsigned int v4u;
typedef __attribute__((ext_vector_type(2)))  unsigned int v2u;
typedef __attribute__((ext_vector_type(8)))  int v8i;
typedef __attribute__((ext_vector_type(4)))  int v4i;

union Frag16 { v16bf v; u16_t u[16]; v4u q[2]; };
union Pack8  { v4u v; u16_t s[8]; };

__device__ __forceinline__ u16_t f2bf(float f) {
  unsigned int u = __float_as_uint(f);
  u += 0x7FFFu + ((u >> 16) & 1u);          // round-to-nearest-even
  return (u16_t)(u >> 16);
}

// ---------------- TDM: 2-D tensor_load_to_lds (bf16 elements) ----------------
// D# per CDNA5 ISA. OOB rows/cols read as zero (free edge
// handling). pad: interval 3 = every 16 DWORDs (one 32-elem bf16 row),
// amount 3 = 4 DWORDs (16B) -> reproduces the 80B padded LDS pitch.
__device__ __forceinline__ void tdm_load_2d_bf16(
    unsigned lds_off, const void* gaddr,
    unsigned tensor_d0, unsigned tensor_d1,
    unsigned long stride0_elems,
    unsigned tile_d0, unsigned tile_d1,
    bool pad_en, unsigned pad_interval_code, unsigned pad_amount_code)
{
  unsigned long ga = (unsigned long)gaddr;
  v4u g0;
  g0.x = 1u;                                                // count=1, user mode
  g0.y = lds_off;                                           // lds_addr (bytes)
  g0.z = (unsigned)(ga & 0xFFFFFFFFu);                      // global_addr[31:0]
  g0.w = (unsigned)((ga >> 32) & 0x01FFFFFFu) | (2u << 30); // addr[56:32] | type=2
  unsigned w0 = (1u << 16)                                  // data_size = 2 bytes
              | (pad_en ? (1u << 20) : 0u)
              | (pad_interval_code << 22)
              | (pad_amount_code << 25);
  unsigned w1 = (tensor_d0 & 0xFFFFu) << 16;                // tensor_dim0 @ bits 79:48
  unsigned w2 = (tensor_d0 >> 16) | ((tensor_d1 & 0xFFFFu) << 16);
  unsigned w3 = (tensor_d1 >> 16) | (tile_d0 << 16);        // tile_dim0 @ bits 127:112
  unsigned w4 = (tile_d1 & 0xFFFFu);                        // tile_dim1; tile_dim2=0
  unsigned w5 = (unsigned)(stride0_elems & 0xFFFFFFFFu);    // dim0_stride lo32
  unsigned w6 = (unsigned)((stride0_elems >> 32) & 0xFFFFu);// stride hi16; dim1_stride=0
  unsigned w7 = 0;
  v8i g1; g1[0]=(int)w0; g1[1]=(int)w1; g1[2]=(int)w2; g1[3]=(int)w3;
          g1[4]=(int)w4; g1[5]=(int)w5; g1[6]=(int)w6; g1[7]=(int)w7;
  v4i gz = {0, 0, 0, 0};
#if __clang_major__ >= 23
  v8i gz8 = {0, 0, 0, 0, 0, 0, 0, 0};
  __builtin_amdgcn_tensor_load_to_lds(g0, g1, gz, gz, gz8, 0);
#else
  __builtin_amdgcn_tensor_load_to_lds(g0, g1, gz, gz, 0);
#endif
}

// ---------------- generic bf16 WMMA GEMM (double-buffered, TDM A-path) -------
// C[M,N] = A[M,K](bf16 rm) * B[K,N](bf16 rm) + bias + epilogue
// EPI: 0 = bias only (f32 out)          1 = bias + exact GELU (bf16 out)
//      2 = bias + residual in-place f32 3 = bias + pos-embed add (f32 out, ld 768)
// Pipeline per K-step: fill B(k)->Bs[cur]; wait TDM(k); barrier;
//                      issue TDM(k+1)->As[1-cur] (overlaps compute); compute(k).
#define BM 128
#define BN 128
#define BK 32
#define LDS_P 40   // 32 + 8 pad -> 80B pitch (16B aligned)

template<int EPI, bool TDMA>
__global__ __launch_bounds__(256) void k_gemm_bf16(
    const u16_t* __restrict__ A, int lda, long sA,
    const u16_t* __restrict__ Bw, int ldb, long sB,
    const float* __restrict__ bias,
    void* __restrict__ Cv, int ldc, long sC,
    const float* __restrict__ pos,
    int M, int N, int K)
{
  __shared__ __align__(16) u16_t As[2][BM * LDS_P];   // row-major [row][k]
  __shared__ __align__(16) u16_t BsT[2][BN * LDS_P];  // transposed [col][k]

  const int tid   = threadIdx.x;
  const int lane  = tid & 31;
  const int wave  = tid >> 5;           // 8 waves
  const int half  = lane >> 4;          // ISA 7.12.2 half-wave split
  const int rr    = lane & 15;
  const int m0    = (wave >> 1) * 32;   // 4 waves along M (32 rows each)
  const int n0    = (wave &  1) * 64;   // 2 waves along N (64 cols each)
  const int mBase = blockIdx.y * BM;
  const int nBase = blockIdx.x * BN;
  const int z     = blockIdx.z;

  A  += (long)z * sA;
  Bw += (long)z * sB;
  const long cOff = (long)z * sC;
  float* Cf = (float*)Cv;
  u16_t* Ch = (u16_t*)Cv;

  const v8f vzero = {0.f,0.f,0.f,0.f,0.f,0.f,0.f,0.f};
  v8f acc[2][4] = {{vzero, vzero, vzero, vzero}, {vzero, vzero, vzero, vzero}};

  const int ksteps = (K + BK - 1) / BK;

  auto issueA = [&](int kt, int buf) {
    if constexpr (TDMA) {
      if (wave == 0) {   // one TDM op per workgroup; EXEC ignored by TDM
        tdm_load_2d_bf16((unsigned)(unsigned long)(void*)&As[buf][0],
                         A + (long)mBase * lda + kt * BK,
                         (unsigned)(K - kt * BK), (unsigned)(M - mBase),
                         (unsigned long)lda, BK, BM,
                         /*pad*/true, /*every 16 DW*/3, /*4 DW*/3);
      }
    } else {
#pragma unroll
      for (int it = 0; it < 2; ++it) {
        int i  = tid + it * 256;
        int r  = i >> 2;
        int cb = (i & 3) * 8;
        int gr = mBase + r, gc = kt * BK + cb;
        v4u val = {0u, 0u, 0u, 0u};
        if (gr < M && gc < K) val = *(const v4u*)(A + (long)gr * lda + gc);
        *(v4u*)&As[buf][r * LDS_P + cb] = val;
      }
    }
  };

  issueA(0, 0);

  for (int kt = 0; kt < ksteps; ++kt) {
    const int cur = kt & 1;
    const int k0  = kt * BK;

    // B tile (BK x BN): two 16B global loads per thread, stored transposed.
    // Writes Bs[cur]; last readers were compute(kt-2), fenced by barrier(kt-1).
#pragma unroll
    for (int it = 0; it < 2; ++it) {
      int i  = tid + it * 256;   // 512 runs of 8
      int r  = i >> 4;           // k row 0..31
      int cb = (i & 15) * 8;     // col run
      int gr = k0 + r, gc = nBase + cb;
      Pack8 pb; pb.v = (v4u){0u, 0u, 0u, 0u};
      if (gr < K && gc < N) pb.v = *(const v4u*)(Bw + (long)gr * ldb + gc);
      if (gr + BK < K) __builtin_prefetch(Bw + (long)(gr + BK) * ldb + gc, 0, 1);
#pragma unroll
      for (int j = 0; j < 8; ++j) BsT[cur][(cb + j) * LDS_P + r] = pb.s[j];
    }

    if constexpr (TDMA) {
      if (wave == 0) __builtin_amdgcn_s_wait_tensorcnt(0);   // TDM(kt) landed
    }
    __syncthreads();

    // overlap: kick TDM for the next A tile while this step computes
    if (kt + 1 < ksteps) issueA(kt + 1, 1 - cur);

    const u16_t* Asp = &As[cur][0];
    const u16_t* Bsp = &BsT[cur][0];

    Frag16 fa[2], fb[4];
#pragma unroll
    for (int tm = 0; tm < 2; ++tm) {
      int row = m0 + tm * 16 + rr;
      // A elems 0-7 -> K = half*8+j ; elems 8-15 -> K = 16+half*8+j
      fa[tm].q[0] = *(const v4u*)&Asp[row * LDS_P + half * 8];
      fa[tm].q[1] = *(const v4u*)&Asp[row * LDS_P + 16 + half * 8];
    }
#pragma unroll
    for (int tn = 0; tn < 4; ++tn) {
      int col = n0 + tn * 16 + rr;
      // B elems 0-7 -> K = half*8+j ; elems 8-15 -> K = 16+half*8+j
      fb[tn].q[0] = *(const v4u*)&Bsp[col * LDS_P + half * 8];
      fb[tn].q[1] = *(const v4u*)&Bsp[col * LDS_P + 16 + half * 8];
    }

#pragma unroll
    for (int tm = 0; tm < 2; ++tm)
#pragma unroll
      for (int tn = 0; tn < 4; ++tn)
        acc[tm][tn] = __builtin_amdgcn_wmma_f32_16x16x32_bf16(
            false, fa[tm].v, false, fb[tn].v, (short)0, acc[tm][tn], false, false);
    __syncthreads();
  }

  // store: D layout -> VGPR i holds M = half*8 + i, N = lane&15
#pragma unroll
  for (int tm = 0; tm < 2; ++tm) {
#pragma unroll
    for (int tn = 0; tn < 4; ++tn) {
      int col = nBase + n0 + tn * 16 + rr;
#pragma unroll
      for (int i = 0; i < 8; ++i) {
        int row = mBase + m0 + tm * 16 + half * 8 + i;
        if (row < M && col < N) {
          float v = acc[tm][tn][i];
          if (bias) v += bias[col];
          long idx = cOff + (long)row * ldc + col;
          if constexpr (EPI == 1) {
            v = 0.5f * v * (1.0f + erff(v * 0.70710678118f));   // exact GELU
            Ch[idx] = f2bf(v);
          } else if constexpr (EPI == 2) {
            Cf[idx] = Cf[idx] + v;                              // residual in-place
          } else if constexpr (EPI == 3) {
            Cf[idx] = v + pos[(long)row * 768 + col];           // pos-embed add
          } else {
            Cf[idx] = v;
          }
        }
      }
    }
  }
}

// ---------------- small helper kernels ----------------
__global__ void k_cvt4(const float* __restrict__ src, u16_t* __restrict__ dst, long n) {
  long i = ((long)blockIdx.x * blockDim.x + threadIdx.x) * 4;
  if (i >= n) return;
  v4f f = *(const v4f*)(src + i);
  v2u o;
  o.x = (unsigned)f2bf(f.x) | ((unsigned)f2bf(f.y) << 16);
  o.y = (unsigned)f2bf(f.z) | ((unsigned)f2bf(f.w) << 16);
  *(v2u*)(dst + i) = o;
}

// conv_w [D, CIN*P*P=768] -> transposed bf16 [768, D]
__global__ void k_convw(const float* __restrict__ w, u16_t* __restrict__ wt) {
  long i = (long)blockIdx.x * blockDim.x + threadIdx.x;
  if (i >= (long)D_ * D_) return;
  int d = (int)(i % D_);
  int k = (int)(i / D_);
  wt[(long)k * D_ + d] = f2bf(w[(long)d * D_ + k]);
}

// im2col: x[B,3,224,224] -> Apat bf16 [B,196,768] in (ci,iy,ix) order
__global__ void k_im2col(const float* __restrict__ x, u16_t* __restrict__ apat) {
  long i = (long)blockIdx.x * blockDim.x + threadIdx.x;
  long total = (long)B_ * NPATCH * D_;
  if (i >= total) return;
  int c = (int)(i % D_);
  long t = i / D_;
  int p = (int)(t % NPATCH);
  int b = (int)(t / NPATCH);
  int ci = c >> 8, rem = c & 255, iy = rem >> 4, ix = rem & 15;
  int py = p / 14, px = p % 14;
  apat[i] = f2bf(x[(((long)b * 3 + ci) * 224 + (py * 16 + iy)) * 224 + (px * 16 + ix)]);
}

// token 0 = cls + pos[0]; pad rows 197..207 zeroed
__global__ void k_tokens(float* __restrict__ h, const float* __restrict__ cls,
                         const float* __restrict__ pos) {
  long i = (long)blockIdx.x * blockDim.x + threadIdx.x;
  long total = (long)B_ * 12 * D_;
  if (i >= total) return;
  int c = (int)(i % D_);
  int r = (int)((i / D_) % 12);
  int b = (int)(i / ((long)12 * D_));
  long row = (r == 0) ? 0 : (long)(NPATCH + r);
  float v = (r == 0) ? (cls[c] + pos[c]) : 0.f;
  h[((long)b * NPAD + row) * D_ + c] = v;
}

__global__ __launch_bounds__(256) void k_layernorm(
    const float* __restrict__ h, const float* __restrict__ w,
    const float* __restrict__ b, u16_t* __restrict__ out) {
  __shared__ float red[256];
  int row = blockIdx.x;
  int n = row % NPAD;
  const float* xr = h + (long)row * D_;
  int tid = threadIdx.x;
  float s = 0.f;
  for (int i = tid; i < D_; i += 256) s += xr[i];
  red[tid] = s; __syncthreads();
  for (int st = 128; st > 0; st >>= 1) { if (tid < st) red[tid] += red[tid + st]; __syncthreads(); }
  float mean = red[0] * (1.0f / D_); __syncthreads();
  float vs = 0.f;
  for (int i = tid; i < D_; i += 256) { float d = xr[i] - mean; vs += d * d; }
  red[tid] = vs; __syncthreads();
  for (int st = 128; st > 0; st >>= 1) { if (tid < st) red[tid] += red[tid + st]; __syncthreads(); }
  float inv = rsqrtf(red[0] * (1.0f / D_) + 1e-6f);
  for (int i = tid; i < D_; i += 256) {
    float y = (xr[i] - mean) * inv * w[i] + b[i];
    out[(long)row * D_ + i] = (n >= NTOK) ? (u16_t)0 : f2bf(y);
  }
}

// qkvf f32 [B,NPAD,2304] -> qh bf16 (pre-scaled), kt bf16 (transposed), vh bf16
__global__ void k_qkvsplit(const float* __restrict__ qkvf, u16_t* __restrict__ qh,
                           u16_t* __restrict__ kt, u16_t* __restrict__ vh) {
  long i = (long)blockIdx.x * blockDim.x + threadIdx.x;
  long total = (long)B_ * H_ * NPAD * DH_;
  if (i >= total) return;
  int d = (int)(i & 63);
  long t = i >> 6;
  int n = (int)(t % NPAD); t /= NPAD;
  int hh = (int)(t % H_);
  int b = (int)(t / H_);
  long base = ((long)b * NPAD + n) * (3 * D_) + (long)hh * DH_ + d;
  float q = qkvf[base];
  float k = qkvf[base + D_];
  float v = qkvf[base + 2 * D_];
  long bh = (long)b * H_ + hh;
  qh[(bh * NPAD + n) * DH_ + d] = f2bf(q * 0.125f);
  kt[(bh * DH_ + d) * NPAD + n] = f2bf(k);
  vh[(bh * NPAD + n) * DH_ + d] = f2bf(v);
}

__global__ __launch_bounds__(256) void k_softmax(const float* __restrict__ s,
                                                 u16_t* __restrict__ p) {
  __shared__ float red[256];
  long row = blockIdx.x;
  const float* sr = s + row * NPAD;
  int tid = threadIdx.x;
  float x = (tid < NTOK) ? sr[tid] : -3.0e38f;
  red[tid] = x; __syncthreads();
  for (int st = 128; st > 0; st >>= 1) { if (tid < st) red[tid] = fmaxf(red[tid], red[tid + st]); __syncthreads(); }
  float m = red[0]; __syncthreads();
  float e = (tid < NTOK) ? __expf(x - m) : 0.f;
  red[tid] = e; __syncthreads();
  for (int st = 128; st > 0; st >>= 1) { if (tid < st) red[tid] += red[tid + st]; __syncthreads(); }
  float inv = 1.0f / red[0];
  if (tid < NPAD) p[row * NPAD + tid] = f2bf(e * inv);
}

// of f32 [B*H,NPAD,64] -> obf bf16 [B,NPAD,768]
__global__ void k_orepack(const float* __restrict__ of, u16_t* __restrict__ obf) {
  long i = (long)blockIdx.x * blockDim.x + threadIdx.x;
  long total = (long)B_ * NPAD * D_;
  if (i >= total) return;
  int c = (int)(i % D_);
  long t = i / D_;
  int n = (int)(t % NPAD);
  int b = (int)(t / NPAD);
  int hh = c >> 6, d = c & 63;
  obf[i] = f2bf(of[(((long)b * H_ + hh) * NPAD + n) * DH_ + d]);
}

__global__ __launch_bounds__(256) void k_final(const float* __restrict__ h,
                                               const float* __restrict__ w,
                                               const float* __restrict__ b,
                                               float* __restrict__ out) {
  __shared__ float red[256];
  int bi = blockIdx.x;
  const float* xr = h + (long)bi * NPAD * D_;
  int tid = threadIdx.x;
  float s = 0.f;
  for (int i = tid; i < D_; i += 256) s += xr[i];
  red[tid] = s; __syncthreads();
  for (int st = 128; st > 0; st >>= 1) { if (tid < st) red[tid] += red[tid + st]; __syncthreads(); }
  float mean = red[0] * (1.0f / D_); __syncthreads();
  float vs = 0.f;
  for (int i = tid; i < D_; i += 256) { float d = xr[i] - mean; vs += d * d; }
  red[tid] = vs; __syncthreads();
  for (int st = 128; st > 0; st >>= 1) { if (tid < st) red[tid] += red[tid + st]; __syncthreads(); }
  float inv = rsqrtf(red[0] * (1.0f / D_) + 1e-6f);
  for (int i = tid; i < D_; i += 256)
    out[(long)bi * D_ + i] = (xr[i] - mean) * inv * w[i] + b[i];
}

// ---------------- orchestration ----------------
extern "C" void kernel_launch(void* const* d_in, const int* in_sizes, int n_in,
                              void* d_out, int out_size, void* d_ws, size_t ws_size,
                              hipStream_t stream) {
  (void)in_sizes; (void)n_in; (void)out_size; (void)ws_size;
  const float* x      = (const float*)d_in[0];
  const float* conv_w = (const float*)d_in[1];
  const float* conv_b = (const float*)d_in[2];
  const float* cls    = (const float*)d_in[3];
  const float* pos    = (const float*)d_in[4];
  const float* ln1w   = (const float*)d_in[5];
  const float* ln1b   = (const float*)d_in[6];
  const float* qkvw   = (const float*)d_in[7];
  const float* qkvb   = (const float*)d_in[8];
  const float* projw  = (const float*)d_in[9];
  const float* projb  = (const float*)d_in[10];
  const float* ln2w   = (const float*)d_in[11];
  const float* ln2b   = (const float*)d_in[12];
  const float* fc1w   = (const float*)d_in[13];
  const float* fc1b   = (const float*)d_in[14];
  const float* fc2w   = (const float*)d_in[15];
  const float* fc2b   = (const float*)d_in[16];
  const float* lnfw   = (const float*)d_in[17];
  const float* lnfb   = (const float*)d_in[18];
  float* out = (float*)d_out;

  char* base = (char*)d_ws;
  size_t off = 0;
  auto alloc = [&](size_t bytes) -> char* {
    char* p = base + off;
    off = (off + bytes + 255) & ~(size_t)255;
    return p;
  };
  float* h      = (float*)alloc((size_t)B_ * NPAD * D_ * 4);
  float* qkvf   = (float*)alloc((size_t)B_ * NPAD * 3 * D_ * 4);
  float* scores = (float*)alloc((size_t)B_ * H_ * NPAD * NPAD * 4);
  float* of     = (float*)alloc((size_t)B_ * H_ * NPAD * DH_ * 4);
  u16_t* abuf   = (u16_t*)alloc((size_t)B_ * NPAD * D_ * 2);
  u16_t* gbuf   = (u16_t*)alloc((size_t)B_ * NPAD * F_ * 2);
  u16_t* probs  = (u16_t*)alloc((size_t)B_ * H_ * NPAD * NPAD * 2);
  u16_t* qh     = (u16_t*)alloc((size_t)B_ * H_ * NPAD * DH_ * 2);
  u16_t* kt     = (u16_t*)alloc((size_t)B_ * H_ * DH_ * NPAD * 2);
  u16_t* vh     = (u16_t*)alloc((size_t)B_ * H_ * NPAD * DH_ * 2);
  u16_t* wq     = (u16_t*)alloc((size_t)D_ * 3 * D_ * 2);
  u16_t* wp     = (u16_t*)alloc((size_t)D_ * D_ * 2);
  u16_t* w1     = (u16_t*)alloc((size_t)D_ * F_ * 2);
  u16_t* w2     = (u16_t*)alloc((size_t)F_ * D_ * 2);
  u16_t* wconv  = (u16_t*)alloc((size_t)D_ * D_ * 2);
  u16_t* apat   = abuf;   // reuse: im2col buffer dead after patch GEMM

  auto cdiv = [](long a, long b) { return (int)((a + b - 1) / b); };

  // patch embedding
  k_convw <<<cdiv((long)D_ * D_, 256), 256, 0, stream>>>(conv_w, wconv);
  k_im2col<<<cdiv((long)B_ * NPATCH * D_, 256), 256, 0, stream>>>(x, apat);
  k_tokens<<<cdiv((long)B_ * 12 * D_, 256), 256, 0, stream>>>(h, cls, pos);
  {
    dim3 g(cdiv(D_, BN), cdiv(NPATCH, BM), B_);
    k_gemm_bf16<3, true><<<g, 256, 0, stream>>>(apat, D_, (long)NPATCH * D_,
                                                wconv, D_, 0, conv_b,
                                                h + D_, D_, (long)NPAD * D_,
                                                pos + D_, NPATCH, D_, D_);
  }

  for (int l = 0; l < L_; ++l) {
    k_cvt4<<<cdiv((long)D_ * 3 * D_ / 4, 256), 256, 0, stream>>>(qkvw + (long)l * D_ * 3 * D_, wq, (long)D_ * 3 * D_);
    k_cvt4<<<cdiv((long)D_ * D_ / 4, 256), 256, 0, stream>>>(projw + (long)l * D_ * D_, wp, (long)D_ * D_);
    k_cvt4<<<cdiv((long)D_ * F_ / 4, 256), 256, 0, stream>>>(fc1w + (long)l * D_ * F_, w1, (long)D_ * F_);
    k_cvt4<<<cdiv((long)F_ * D_ / 4, 256), 256, 0, stream>>>(fc2w + (long)l * F_ * D_, w2, (long)F_ * D_);

    k_layernorm<<<B_ * NPAD, 256, 0, stream>>>(h, ln1w + (long)l * D_, ln1b + (long)l * D_, abuf);

    {  // QKV: [NPAD,768] x [768,2304]
      dim3 g(cdiv(3 * D_, BN), cdiv(NPAD, BM), B_);
      k_gemm_bf16<0, true><<<g, 256, 0, stream>>>(abuf, D_, (long)NPAD * D_,
                                                  wq, 3 * D_, 0, qkvb + (long)l * 3 * D_,
                                                  qkvf, 3 * D_, (long)NPAD * 3 * D_,
                                                  nullptr, NPAD, 3 * D_, D_);
    }
    k_qkvsplit<<<cdiv((long)B_ * H_ * NPAD * DH_, 256), 256, 0, stream>>>(qkvf, qh, kt, vh);

    {  // scores = (q*scale) @ k^T : [NPAD,64] x [64,NPAD], 384 batches
      dim3 g(cdiv(NPAD, BN), cdiv(NPAD, BM), B_ * H_);
      k_gemm_bf16<0, true><<<g, 256, 0, stream>>>(qh, DH_, (long)NPAD * DH_,
                                                  kt, NPAD, (long)DH_ * NPAD, nullptr,
                                                  scores, NPAD, (long)NPAD * NPAD,
                                                  nullptr, NPAD, NPAD, DH_);
    }
    k_softmax<<<B_ * H_ * NPAD, 256, 0, stream>>>(scores, probs);

    {  // o = probs @ v : [NPAD,NPAD] x [NPAD,64]
      dim3 g(cdiv(DH_, BN), cdiv(NPAD, BM), B_ * H_);
      k_gemm_bf16<0, true><<<g, 256, 0, stream>>>(probs, NPAD, (long)NPAD * NPAD,
                                                  vh, DH_, (long)NPAD * DH_, nullptr,
                                                  of, DH_, (long)NPAD * DH_,
                                                  nullptr, NPAD, DH_, NPAD);
    }
    k_orepack<<<cdiv((long)B_ * NPAD * D_, 256), 256, 0, stream>>>(of, abuf);

    {  // proj + residual (in place into h)
      dim3 g(cdiv(D_, BN), cdiv(NPAD, BM), B_);
      k_gemm_bf16<2, true><<<g, 256, 0, stream>>>(abuf, D_, (long)NPAD * D_,
                                                  wp, D_, 0, projb + (long)l * D_,
                                                  h, D_, (long)NPAD * D_,
                                                  nullptr, NPAD, D_, D_);
    }

    k_layernorm<<<B_ * NPAD, 256, 0, stream>>>(h, ln2w + (long)l * D_, ln2b + (long)l * D_, abuf);
    {  // FC1 + GELU (bf16 out)
      dim3 g(cdiv(F_, BN), cdiv(NPAD, BM), B_);
      k_gemm_bf16<1, true><<<g, 256, 0, stream>>>(abuf, D_, (long)NPAD * D_,
                                                  w1, F_, 0, fc1b + (long)l * F_,
                                                  gbuf, F_, (long)NPAD * F_,
                                                  nullptr, NPAD, F_, D_);
    }
    {  // FC2 + residual
      dim3 g(cdiv(D_, BN), cdiv(NPAD, BM), B_);
      k_gemm_bf16<2, true><<<g, 256, 0, stream>>>(gbuf, F_, (long)NPAD * F_,
                                                  w2, D_, 0, fc2b + (long)l * D_,
                                                  h, D_, (long)NPAD * D_,
                                                  nullptr, NPAD, D_, F_);
    }
  }

  k_final<<<B_, 256, 0, stream>>>(h, lnfw, lnfb, out);
}
